// DSConv2d_56401510531643
// MI455X (gfx1250) — matrix operand order repaired and checked by reading, not compiled, not run on hardware
//
#include <hip/hip_runtime.h>
#include <hip/hip_bf16.h>
#include <math.h>

typedef __bf16  v16bf __attribute__((ext_vector_type(16)));
typedef float   v8f   __attribute__((ext_vector_type(8)));

#define HW      6400      // 80*80
#define Wd      80
#define C1      128
#define C2      128
#define NTAP    9
#define KTOT    (C1 * NTAP)          // 1152
#define KCHUNKS (KTOT / 32)          // 36
#define MTILES  (C2 / 16)            // 8
#define OFFCH   18
#define NB      64                   // pixels per workgroup tile
#define NT      (NB / 16)            // 4 N-tiles
// LDS B tile: element (k, n) -> (k>>4)*(NB*16) + n*16 + (k&15)   (bf16)
#define LDSH    ((KTOT / 16) * NB * 16)   // 73728 halves = 144 KB

// ---------------------------------------------------------------------------
// Kernel A: swizzle fp32 weights [M, KTOT] into bf16 WMMA A-fragment layout
// [mtiles][kc][lane][16 halves];   m = lane%16,
// half j -> k = kc*32 + (lane/16)*16 + j.
// Rows m >= mrows are zero-padded (used for the 18-channel offset conv).
// ---------------------------------------------------------------------------
__global__ void prep_weights_kernel(const float* __restrict__ wsrc,
                                    __bf16* __restrict__ wdst,
                                    int mtiles, int mrows) {
    int idx = blockIdx.x * blockDim.x + threadIdx.x;
    if (idx >= mtiles * KCHUNKS * 32 * 16) return;
    int j    = idx & 15;
    int lane = (idx >> 4) & 31;
    int kc   = (idx >> 9) % KCHUNKS;
    int mt   = idx / (512 * KCHUNKS);
    int m = mt * 16 + (lane & 15);
    int k = kc * 32 + ((lane >> 4) << 4) + j;
    float v = (m < mrows) ? wsrc[(size_t)m * KTOT + k] : 0.0f;
    wdst[idx] = (__bf16)v;
}

// ---------------------------------------------------------------------------
// Kernel B: offset conv as WMMA GEMM.  A = offset_w [32(pad), 1152] bf16,
// B = im2col patch [1152, 64 pixels] built in LDS.  8 waves = 2 Mtiles x 4 Nt.
// ---------------------------------------------------------------------------
__global__ __launch_bounds__(256) void offset_wmma_kernel(
        const float* __restrict__ x,
        const __bf16* __restrict__ owprep,
        const float* __restrict__ ob,
        float* __restrict__ off) {
    __shared__ __bf16 smp[LDSH];

    const int pix = blockIdx.x * NB;
    const int b   = pix / HW;
    const int rem = pix % HW;              // tile never crosses batch (6400%64==0)

    const int t   = threadIdx.x;
    const int n64 = t & 63;                // pixel within tile
    const int g   = t >> 6;                // channel group 0..3 (32 ch each)
    const int pidx = rem + n64;
    const int h = pidx / Wd, w = pidx % Wd;

    const float* xb = x + (size_t)b * C1 * HW;

    // ---- phase 1: im2col patch (zero pad) -> LDS bf16 ----
#pragma unroll
    for (int p = 0; p < NTAP; ++p) {
        int y  = h - 1 + p / 3;
        int xx = w - 1 + p % 3;
        bool ok = ((unsigned)y < (unsigned)Wd) && ((unsigned)xx < (unsigned)Wd);
        int src = y * Wd + xx;
#pragma unroll 4
        for (int cc = 0; cc < 32; ++cc) {
            int ci = g * 32 + cc;
            float v = ok ? xb[(size_t)ci * HW + src] : 0.0f;
            int k = ci * NTAP + p;
            smp[((k >> 4) * (NB * 16)) + (n64 << 4) + (k & 15)] = (__bf16)v;
        }
    }
    __syncthreads();

    // ---- phase 2: WMMA ----
    const int lane = t & 31;
    const int wave = t >> 5;
    const int mt = wave >> 2;              // 0..1
    const int nt = wave & 3;               // 0..3
    const int lblk = lane >> 4, ln = lane & 15;
    v8f acc = {};
    const v16bf* aptr = (const v16bf*)owprep + ((size_t)mt * KCHUNKS) * 32 + lane;
#pragma unroll 4
    for (int kc = 0; kc < KCHUNKS; ++kc) {
        v16bf a = aptr[kc * 32];
        const v16bf* bptr = (const v16bf*)
            &smp[((kc * 2 + lblk) * (NB * 16)) + (((nt << 4) + ln) << 4)];
        acc = __builtin_amdgcn_wmma_f32_16x16x32_bf16(
                  false, a, false, *bptr, (short)0, acc, false, false);
    }

    // ---- store rows co < 18 ----
    const int opix = rem + nt * 16 + ln;
#pragma unroll
    for (int r = 0; r < 8; ++r) {
        int co = mt * 16 + r + (lblk << 3);
        if (co < OFFCH)
            off[((size_t)b * OFFCH + co) * HW + opix] = acc[r] + ob[co];
    }
}

// ---------------------------------------------------------------------------
// Kernel C: deformable bilinear sampling into LDS (bf16, B-fragment layout)
// + WMMA contraction.  8 waves = 8 M-tiles; each wave does 4 N-tiles,
// reusing every A fragment 4x (cuts weight L2 traffic 4x).
// ---------------------------------------------------------------------------
__global__ __launch_bounds__(256) void deform_wmma_kernel(
        const float* __restrict__ x,
        const float* __restrict__ off,
        const __bf16* __restrict__ wprep,
        const float* __restrict__ bias,
        float* __restrict__ out) {
    __shared__ __bf16 smp[LDSH];

    const int pix = blockIdx.x * NB;
    const int b   = pix / HW;
    const int rem = pix % HW;

    const int t   = threadIdx.x;
    const int n64 = t & 63;
    const int g   = t >> 6;                // 0..3, 32 channels each
    const int pidx = rem + n64;
    const int h = pidx / Wd, w = pidx % Wd;

    const float* xb  = x + (size_t)b * C1 * HW;
    const float* obp = off + (size_t)b * OFFCH * HW + pidx;

    // ---- phase 1: bilinear sampling -> LDS bf16 ----
    for (int p = 0; p < NTAP; ++p) {
        float dy = obp[(2 * p) * HW];
        float dx = obp[(2 * p + 1) * HW];
        float py = (float)(h - 1 + p / 3) + dy;
        float px = (float)(w - 1 + p % 3) + dx;
        float y0f = floorf(py), x0f = floorf(px);
        int   y0 = (int)y0f,   x0i = (int)x0f;
        int   y1 = y0 + 1,     x1i = x0i + 1;
        float wy1 = py - y0f, wy0 = 1.0f - wy1;
        float wx1 = px - x0f, wx0 = 1.0f - wx1;
        float vy0 = ((unsigned)y0  < (unsigned)Wd) ? 1.0f : 0.0f;
        float vy1 = ((unsigned)y1  < (unsigned)Wd) ? 1.0f : 0.0f;
        float vx0 = ((unsigned)x0i < (unsigned)Wd) ? 1.0f : 0.0f;
        float vx1 = ((unsigned)x1i < (unsigned)Wd) ? 1.0f : 0.0f;
        int yc0 = min(max(y0, 0), Wd - 1), yc1 = min(max(y1, 0), Wd - 1);
        int xc0 = min(max(x0i, 0), Wd - 1), xc1 = min(max(x1i, 0), Wd - 1);
        float m00 = wy0 * wx0 * vy0 * vx0;
        float m01 = wy0 * wx1 * vy0 * vx1;
        float m10 = wy1 * wx0 * vy1 * vx0;
        float m11 = wy1 * wx1 * vy1 * vx1;
        int o00 = yc0 * Wd + xc0, o01 = yc0 * Wd + xc1;
        int o10 = yc1 * Wd + xc0, o11 = yc1 * Wd + xc1;
#pragma unroll 4
        for (int cc = 0; cc < 32; ++cc) {
            int ci = g * 32 + cc;
            const float* xc = xb + (size_t)ci * HW;
            float v = m00 * xc[o00] + m01 * xc[o01]
                    + m10 * xc[o10] + m11 * xc[o11];
            int k = ci * NTAP + p;
            smp[((k >> 4) * (NB * 16)) + (n64 << 4) + (k & 15)] = (__bf16)v;
        }
    }
    __syncthreads();

    // ---- phase 2: WMMA, wave = M-tile, 4 N-tiles per wave ----
    const int lane = t & 31;
    const int wave = t >> 5;
    const int lblk = lane >> 4, ln = lane & 15;
    v8f acc0 = {}, acc1 = {}, acc2 = {}, acc3 = {};
    const v16bf* aptr = (const v16bf*)wprep + ((size_t)wave * KCHUNKS) * 32 + lane;
#pragma unroll 2
    for (int kc = 0; kc < KCHUNKS; ++kc) {
        v16bf a = aptr[kc * 32];
        const __bf16* brow = &smp[((kc * 2 + lblk) * (NB * 16)) + (ln << 4)];
        v16bf b0 = *(const v16bf*)(brow);
        v16bf b1 = *(const v16bf*)(brow + 256);
        v16bf b2 = *(const v16bf*)(brow + 512);
        v16bf b3 = *(const v16bf*)(brow + 768);
        acc0 = __builtin_amdgcn_wmma_f32_16x16x32_bf16(false, a, false, b0, (short)0, acc0, false, false);
        acc1 = __builtin_amdgcn_wmma_f32_16x16x32_bf16(false, a, false, b1, (short)0, acc1, false, false);
        acc2 = __builtin_amdgcn_wmma_f32_16x16x32_bf16(false, a, false, b2, (short)0, acc2, false, false);
        acc3 = __builtin_amdgcn_wmma_f32_16x16x32_bf16(false, a, false, b3, (short)0, acc3, false, false);
    }

    // ---- store: VGPR r -> M = r + 8*(lane/16), N = lane%16 ----
    float* op = out + ((size_t)b * C2 + wave * 16) * HW + rem + ln;
    v8f accs[4] = {acc0, acc1, acc2, acc3};
#pragma unroll
    for (int nt = 0; nt < NT; ++nt) {
#pragma unroll
        for (int r = 0; r < 8; ++r) {
            int mm = r + (lblk << 3);
            op[(size_t)mm * HW + nt * 16] = accs[nt][r] + bias[wave * 16 + mm];
        }
    }
}

// ---------------------------------------------------------------------------
extern "C" void kernel_launch(void* const* d_in, const int* in_sizes, int n_in,
                              void* d_out, int out_size, void* d_ws, size_t ws_size,
                              hipStream_t stream) {
    const float* x        = (const float*)d_in[0];
    const float* offset_w = (const float*)d_in[1];
    const float* offset_b = (const float*)d_in[2];
    const float* weight   = (const float*)d_in[3];
    const float* bias     = (const float*)d_in[4];
    float* out = (float*)d_out;

    // workspace layout
    char* ws = (char*)d_ws;
    float*  off    = (float*)ws;                                   // 7,372,800 B
    __bf16* wprep  = (__bf16*)(ws + (size_t)16 * OFFCH * HW * 4);  // 294,912 B
    __bf16* owprep = (__bf16*)(ws + (size_t)16 * OFFCH * HW * 4 + MTILES * KCHUNKS * 512 * 2); // 73,728 B

    {
        int n = MTILES * KCHUNKS * 512;     // 147456 : main weights, 8 M-tiles
        prep_weights_kernel<<<(n + 255) / 256, 256, 0, stream>>>(weight, wprep, MTILES, C2);
        int n2 = 2 * KCHUNKS * 512;         // 36864 : offset weights, 2 M-tiles (pad 18->32)
        prep_weights_kernel<<<(n2 + 255) / 256, 256, 0, stream>>>(offset_w, owprep, 2, OFFCH);
    }
    {
        int ntiles = 16 * HW / NB;          // 1600 workgroups
        offset_wmma_kernel<<<ntiles, 256, 0, stream>>>(x, owprep, offset_b, off);
        deform_wmma_kernel<<<ntiles, 256, 0, stream>>>(x, off, wprep, bias, out);
    }
}
